// GatedResidualCondConv_45595372815053
// MI455X (gfx1250) — matrix operand order repaired and checked by reading, not compiled, not run on hardware
//
#include <hip/hip_runtime.h>
#include <hip/hip_bf16.h>
#include <math.h>

typedef __bf16 bf16_t;
typedef __attribute__((ext_vector_type(16))) __bf16 v16bf;
typedef __attribute__((ext_vector_type(8)))  __bf16 v8bf;
typedef __attribute__((ext_vector_type(8)))  float  v8f;

#define NCOND 128
#define NRES  512
#define NDIL  512
#define NSKP  256
#define LEADP 4
#define BATCH 4
#define TLEN  8192
#define TOUT  (TLEN - LEADP)        // 8188
#define TT    64                    // time tile (4 N-tiles of 16)
#define XROWS (TT + LEADP)          // 68 x-rows held (conv tap reach)
#define XP    (NRES + 16)           // 528-half pitch, bank-conflict padding
#define CP    (NCOND + 16)          // 144-half pitch
#define ZP    (NDIL + 16)           // 528-half pitch
#define KT1   36                    // 1152/32 k-steps stage 1
#define KT2   16                    // 512/32 k-steps stage 2
#define MT1   (NDIL / 16)           // 32 m-tiles per matrix, stage 1
#define MT2   ((NRES + NSKP) / 16)  // 48 m-tiles stage 2
#define W1_ELEMS (2 * MT1 * KT1 * 32 * 16)  // 1,179,648 bf16
#define W2_ELEMS (MT2 * KT2 * 32 * 16)      //   393,216 bf16
#define THREADS 512                 // 16 waves (wave32)

#define XS_ELEMS (XROWS * XP)       // 35,904 halves
#define CS_ELEMS (TT * CP)          //  9,216 halves
#define SMEM_ELEMS (XS_ELEMS + CS_ELEMS)  // 45,120 halves = 90,240 B
// z (64 x 528 = 33,792 halves) aliases the x region after a barrier.

__device__ __forceinline__ v8f wmma_bf16(v16bf a, v16bf b, v8f c) {
  return __builtin_amdgcn_wmma_f32_16x16x32_bf16(false, a, false, b,
                                                 (short)0, c, false, false);
}

// K order this code packs into the A fragment (B matches): contiguous, k = 16*(lane/16) + h.
__device__ __forceinline__ int a_klocal(int h, int lane) {
  return h + ((lane >= 16) ? 16 : 0);
}

// ---------------- weight prepack: stage-1 (filt | gate), A-fragment order ---
__global__ void pack_w1(const float* __restrict__ w_sig,
                        const float* __restrict__ w_gate,
                        const float* __restrict__ w_psig,
                        const float* __restrict__ w_pgate,
                        bf16_t* __restrict__ W1) {
  int idx  = blockIdx.x * 256 + threadIdx.x;       // one bf16 per thread
  int h    = idx & 15;
  int lane = (idx >> 4) & 31;
  int tile = idx >> 9;
  int kt   = tile % KT1;
  int tmp  = tile / KT1;
  int mt   = tmp % MT1;
  int mat  = tmp / MT1;                            // 0 = filt, 1 = gate
  int d    = mt * 16 + (lane & 15);
  int k    = kt * 32 + a_klocal(h, lane);
  const float* w  = mat ? w_gate  : w_sig;         // (512, 512, 2)
  const float* wp = mat ? w_pgate : w_psig;        // (512, 128)
  float v;
  if (k < 512)       v = w[(d * 512 + k) * 2 + 0];
  else if (k < 1024) v = w[(d * 512 + (k - 512)) * 2 + 1];
  else               v = wp[d * 128 + (k - 1024)];
  W1[idx] = (bf16_t)v;
}

// ---------------- weight prepack: stage-2 (res | skp) -----------------------
__global__ void pack_w2(const float* __restrict__ w_res,
                        const float* __restrict__ w_skp,
                        bf16_t* __restrict__ W2) {
  int idx  = blockIdx.x * 256 + threadIdx.x;
  int h    = idx & 15;
  int lane = (idx >> 4) & 31;
  int tile = idx >> 9;
  int kt   = tile % KT2;
  int mt   = tile / KT2;                           // 0..47
  int m    = mt * 16 + (lane & 15);
  int k    = kt * 32 + a_klocal(h, lane);
  float v  = (m < NRES) ? w_res[m * 512 + k]
                        : w_skp[(m - NRES) * 512 + k];
  W2[idx] = (bf16_t)v;
}

// One stage-1 k-step: 2 m-tiles x {filt,gate} x 4 n-tiles = 16 WMMAs.
__device__ __forceinline__ void s1_step(int wave, int lane, int kt,
                                        const bf16_t* __restrict__ W1f,
                                        const bf16_t* __restrict__ W1g,
                                        const v16bf b[4],
                                        v8f accF[2][4], v8f accG[2][4]) {
#pragma unroll
  for (int i = 0; i < 2; ++i) {
    int mt = wave * 2 + i;
    size_t off = ((size_t)(mt * KT1 + kt) * 32 + lane) * 16;
    v16bf af = *(const v16bf*)&W1f[off];
    v16bf ag = *(const v16bf*)&W1g[off];
#pragma unroll
    for (int nt = 0; nt < 4; ++nt) accF[i][nt] = wmma_bf16(af, b[nt], accF[i][nt]);
#pragma unroll
    for (int nt = 0; nt < 4; ++nt) accG[i][nt] = wmma_bf16(ag, b[nt], accG[i][nt]);
  }
}

// ---------------- fused main kernel ----------------------------------------
__global__ __launch_bounds__(THREADS)
void wavenet_main(const float* __restrict__ x, const float* __restrict__ cond,
                  const float* __restrict__ b_sig, const float* __restrict__ b_gate,
                  const bf16_t* __restrict__ W1, const bf16_t* __restrict__ W2,
                  float* __restrict__ out_sig, float* __restrict__ out_skp) {
  __shared__ bf16_t smem[SMEM_ELEMS];
  bf16_t* xs  = smem;                 // [XROWS][XP], time-major x tile
  bf16_t* cs  = smem + XS_ELEMS;      // [TT][CP],    time-major cond tile
  bf16_t* zsh = smem;                 // [TT][ZP], aliases xs after barrier

  const int b   = blockIdx.y;
  const int t0  = blockIdx.x * TT;
  const int tid = threadIdx.x;
  const float* xb = x + (size_t)b * NRES * TLEN;

  // --- load x tile (fp32 -> bf16), single copy; taps via row offset ---
  for (int idx = tid; idx < NRES * XROWS; idx += THREADS) {
    int c = idx / XROWS;
    int j = idx % XROWS;
    int t = t0 + j;
    float v = (t < TLEN) ? xb[(size_t)c * TLEN + t] : 0.0f;
    xs[j * XP + c] = (bf16_t)v;
  }
  // --- load cond tile ---
  const float* cb = cond + (size_t)b * NCOND * TLEN;
  for (int idx = tid; idx < NCOND * TT; idx += THREADS) {
    int c = idx / TT;
    int n = idx % TT;
    int t = t0 + n + LEADP;
    float v = (t < TLEN) ? cb[(size_t)c * TLEN + t] : 0.0f;
    cs[n * CP + c] = (bf16_t)v;
  }
  __syncthreads();

  const int lane = tid & 31;
  const int wave = tid >> 5;     // 0..15
  const int nl   = lane & 15;
  const int hi   = lane >> 4;    // 0 or 1
  const int koff = hi * 16;      // B-fragment K offset per half-wave

  // ================= stage 1: filt/gate GEMMs (512x1152 @ 1152x64) =========
  v8f accF[2][4], accG[2][4];
  const v8f vzero = {0.f, 0.f, 0.f, 0.f, 0.f, 0.f, 0.f, 0.f};
#pragma unroll
  for (int i = 0; i < 2; ++i)
#pragma unroll
    for (int nt = 0; nt < 4; ++nt) { accF[i][nt] = vzero; accG[i][nt] = vzero; }

  const bf16_t* W1f = W1;
  const bf16_t* W1g = W1 + (size_t)MT1 * KT1 * 512;

  // tap0: K in [0, 512)
  for (int kt = 0; kt < 16; ++kt) {
    v16bf bfr[4];
#pragma unroll
    for (int nt = 0; nt < 4; ++nt)
      bfr[nt] = *(const v16bf*)&xs[(nt * 16 + nl) * XP + kt * 32 + koff];
    s1_step(wave, lane, kt, W1f, W1g, bfr, accF, accG);
  }
  // tap1: K in [512, 1024) -> same x tile shifted by LEADP rows
  for (int kt = 16; kt < 32; ++kt) {
    v16bf bfr[4];
#pragma unroll
    for (int nt = 0; nt < 4; ++nt)
      bfr[nt] = *(const v16bf*)&xs[(nt * 16 + nl + LEADP) * XP + (kt - 16) * 32 + koff];
    s1_step(wave, lane, kt, W1f, W1g, bfr, accF, accG);
  }
  // cond: K in [1024, 1152)
  for (int kt = 32; kt < 36; ++kt) {
    v16bf bfr[4];
#pragma unroll
    for (int nt = 0; nt < 4; ++nt)
      bfr[nt] = *(const v16bf*)&cs[(nt * 16 + nl) * CP + (kt - 32) * 32 + koff];
    s1_step(wave, lane, kt, W1f, W1g, bfr, accF, accG);
  }

  __syncthreads();   // all waves done reading xs/cs; z may now alias them

  // --- epilogue: bias + tanh*sigmoid -> z in LDS (bf16, time-major) ---
#pragma unroll
  for (int i = 0; i < 2; ++i) {
    int mt = wave * 2 + i;
    int d0 = mt * 16 + hi * 8;   // 8 consecutive d-channels per lane (C/D layout)
    float bs[8], bg[8];
#pragma unroll
    for (int r = 0; r < 8; ++r) { bs[r] = b_sig[d0 + r]; bg[r] = b_gate[d0 + r]; }
#pragma unroll
    for (int nt = 0; nt < 4; ++nt) {
      int n = nt * 16 + nl;
      v8bf zr;
#pragma unroll
      for (int r = 0; r < 8; ++r) {
        float f = accF[i][nt][r] + bs[r];
        float g = accG[i][nt][r] + bg[r];
        float zz = tanhf(f) * (1.0f / (1.0f + __expf(-g)));
        zr[r] = (bf16_t)zz;
      }
      *(v8bf*)&zsh[n * ZP + d0] = zr;
    }
  }
  __syncthreads();

  // ================= stage 2: [res;skp] GEMM (768x512 @ 512x64) ============
  v8f acc2[3][4];
#pragma unroll
  for (int i = 0; i < 3; ++i)
#pragma unroll
    for (int nt = 0; nt < 4; ++nt) acc2[i][nt] = vzero;

  for (int kt = 0; kt < KT2; ++kt) {
    v16bf bfr[4];
#pragma unroll
    for (int nt = 0; nt < 4; ++nt)
      bfr[nt] = *(const v16bf*)&zsh[(nt * 16 + nl) * ZP + kt * 32 + koff];
#pragma unroll
    for (int i = 0; i < 3; ++i) {
      int mo = wave * 3 + i;
      v16bf a = *(const v16bf*)&W2[((size_t)(mo * KT2 + kt) * 32 + lane) * 16];
#pragma unroll
      for (int nt = 0; nt < 4; ++nt) acc2[i][nt] = wmma_bf16(a, bfr[nt], acc2[i][nt]);
    }
  }

  // --- stores: residual add for sig half, guarded partial last tile ---
#pragma unroll
  for (int i = 0; i < 3; ++i) {
    int mo = wave * 3 + i;
    int m0 = mo * 16 + hi * 8;
#pragma unroll
    for (int nt = 0; nt < 4; ++nt) {
      int n = nt * 16 + nl;
      int t = t0 + n;
      if (t < TOUT) {
        if (m0 < NRES) {
#pragma unroll
          for (int r = 0; r < 8; ++r) {
            float v = acc2[i][nt][r] + xb[(size_t)(m0 + r) * TLEN + t + LEADP];
            out_sig[((size_t)b * NRES + m0 + r) * TOUT + t] = v;
          }
        } else {
          int s0 = m0 - NRES;
#pragma unroll
          for (int r = 0; r < 8; ++r)
            out_skp[((size_t)b * NSKP + s0 + r) * TOUT + t] = acc2[i][nt][r];
        }
      }
    }
  }
}

extern "C" void kernel_launch(void* const* d_in, const int* in_sizes, int n_in,
                              void* d_out, int out_size, void* d_ws, size_t ws_size,
                              hipStream_t stream) {
  const float* x       = (const float*)d_in[0];
  const float* cond    = (const float*)d_in[1];
  const float* w_sig   = (const float*)d_in[2];
  const float* b_sig   = (const float*)d_in[3];
  const float* w_gate  = (const float*)d_in[4];
  const float* b_gate  = (const float*)d_in[5];
  const float* w_psig  = (const float*)d_in[6];
  const float* w_pgate = (const float*)d_in[7];
  const float* w_res   = (const float*)d_in[8];
  const float* w_skp   = (const float*)d_in[9];

  bf16_t* W1 = (bf16_t*)d_ws;
  bf16_t* W2 = W1 + W1_ELEMS;
  float* out_sig = (float*)d_out;
  float* out_skp = out_sig + (size_t)BATCH * NRES * TOUT;

  pack_w1<<<W1_ELEMS / 256, 256, 0, stream>>>(w_sig, w_gate, w_psig, w_pgate, W1);
  pack_w2<<<W2_ELEMS / 256, 256, 0, stream>>>(w_res, w_skp, W2);

  dim3 grid((TOUT + TT - 1) / TT, BATCH);   // 128 x 4 workgroups
  wavenet_main<<<grid, THREADS, 0, stream>>>(x, cond, b_sig, b_gate,
                                             W1, W2, out_sig, out_skp);
}
